// GNNClassifier_6227702580040
// MI455X (gfx1250) — matrix operand order repaired and checked by reading, not compiled, not run on hardware
//
#include <hip/hip_runtime.h>

typedef __bf16 bf16_t;
typedef __attribute__((ext_vector_type(16))) __bf16 v16bf;
typedef __attribute__((ext_vector_type(8)))  __bf16 v8bf;
typedef __attribute__((ext_vector_type(8)))  float  v8f;

// Problem constants (match reference)
constexpr int NNODE = 50000;
constexpr int EEDGE = 800000;
constexpr int ETOT  = EEDGE + NNODE;   // with self loops
constexpr int DIN   = 128;
constexpr int HE    = 4;
constexpr int DH    = 64;
constexpr int H4D   = HE * DH;         // 256
constexpr int NCLS  = 6;

// ---- workspace layout (bytes), 1KB aligned, with aliasing across phases ----
constexpr size_t AL(size_t x) { return (x + 1023) & ~size_t(1023); }
constexpr size_t OFF_XB   = 0;                                              // bf16 x      [N,128]
constexpr size_t OFF_W1T  = OFF_XB   + AL((size_t)NNODE * DIN * 2);         // bf16 W1^T   [256,128]
constexpr size_t OFF_W2T  = OFF_W1T  + AL((size_t)H4D * DIN * 2);           // bf16 W2^T   [64,256]
constexpr size_t OFF_H1   = OFF_W2T  + AL((size_t)DH * H4D * 2);            // f32 h1      [N,256]
constexpr size_t OFF_AS1  = OFF_H1   + AL((size_t)NNODE * H4D * 4);         // f32 [N,4]
constexpr size_t OFF_AD1  = OFF_AS1  + AL((size_t)NNODE * HE * 4);
constexpr size_t OFF_M1   = OFF_AD1  + AL((size_t)NNODE * HE * 4);
constexpr size_t OFF_Z1   = OFF_M1   + AL((size_t)NNODE * HE * 4);
constexpr size_t OFF_AGG1 = OFF_Z1   + AL((size_t)NNODE * HE * 4);          // f32 [N,256]
constexpr size_t OFF_AS2  = OFF_AGG1 + AL((size_t)NNODE * H4D * 4);         // f32 [N]
constexpr size_t OFF_AD2  = OFF_AS2  + AL((size_t)NNODE * 4);
constexpr size_t OFF_M2   = OFF_AD2  + AL((size_t)NNODE * 4);
constexpr size_t OFF_Z2   = OFF_M2   + AL((size_t)NNODE * 4);
// aliases over dead regions
constexpr size_t OFF_HB2  = OFF_H1;     // bf16 elu(agg1+b1) [N,256] (h1 dead after aggr1)
constexpr size_t OFF_H2   = OFF_XB;     // f32 h2 [N,64]            (xb dead after gemm1)
constexpr size_t OFF_AGG2 = OFF_AGG1;   // f32 [N,64]               (agg1 dead after cast)

__device__ __forceinline__ void atomicMaxF32(float* addr, float v) {
    if (v >= 0.0f) atomicMax((int*)addr, __float_as_int(v));
    else           atomicMin((unsigned int*)addr, __float_as_uint(v));
}
__device__ __forceinline__ float lrelu02(float v) { return v > 0.0f ? v : 0.2f * v; }
__device__ __forceinline__ float eluf(float v)    { return v > 0.0f ? v : __expf(v) - 1.0f; }

// ---------------- simple element kernels ----------------
__global__ void k_fill(float* p, float v, size_t n) {
    size_t i = (size_t)blockIdx.x * blockDim.x + threadIdx.x;
    size_t stride = (size_t)gridDim.x * blockDim.x;
    for (; i < n; i += stride) p[i] = v;
}

__global__ void k_cast(const float* __restrict__ in, bf16_t* __restrict__ out, size_t n) {
    size_t i = (size_t)blockIdx.x * blockDim.x + threadIdx.x;
    if (i < n) out[i] = (bf16_t)in[i];
}

// out[n*K+k] = bf16(in[k*Nc+n])  (transpose-cast weight for contiguous-K B fragments)
__global__ void k_tcast(const float* __restrict__ in, bf16_t* __restrict__ out, int K, int Nc) {
    int i = blockIdx.x * blockDim.x + threadIdx.x;
    if (i >= K * Nc) return;
    int n = i / K, k = i % K;
    out[(size_t)n * K + k] = (bf16_t)in[(size_t)k * Nc + n];
}

// ---------------- WMMA GEMM: C[M,Nc] = A[M,K] * Bt[Nc,K]^T ----------------
// One wave computes a 16x64 tile (A fragment reused across 4 WMMAs per K step).
template <int K>
__global__ void k_gemm(const bf16_t* __restrict__ A, const bf16_t* __restrict__ Bt,
                       float* __restrict__ C, int M, int Nc) {
    int wave  = blockIdx.x * (blockDim.x >> 5) + (threadIdx.x >> 5);
    int ngrp  = Nc >> 6;                 // groups of 64 columns
    int mt_n  = (M >> 4) * ngrp;
    if (wave >= mt_n) return;            // wave-uniform: EXEC stays all-ones for WMMA
    int mt = wave / ngrp;
    int ng = wave % ngrp;

    int lane = threadIdx.x & 31;
    int l16  = lane & 15;
    int hi   = lane >> 4;                // 0 or 1

    const bf16_t* arow = A + (size_t)(mt * 16 + l16) * K + hi * 16;

    v8f acc[4];
#pragma unroll
    for (int t = 0; t < 4; ++t) acc[t] = (v8f){0.f,0.f,0.f,0.f,0.f,0.f,0.f,0.f};

#pragma unroll
    for (int kb = 0; kb < K; kb += 32) {
        union { v16bf v; v8bf h[2]; } au;
        au.h[0] = *(const v8bf*)(arow + kb);        // K = kb + hi*16 .. +7
        au.h[1] = *(const v8bf*)(arow + kb + 8);    // K = kb+8 + hi*16 .. +7
#pragma unroll
        for (int t = 0; t < 4; ++t) {
            const bf16_t* bp = Bt + (size_t)(ng * 64 + t * 16 + l16) * K + kb + hi * 16;
            v16bf b = *(const v16bf*)bp;            // 16 contiguous K values
            acc[t] = __builtin_amdgcn_wmma_f32_16x16x32_bf16(
                false, au.v, false, b, (short)0, acc[t], false, false);
        }
    }

#pragma unroll
    for (int t = 0; t < 4; ++t) {
        int col = ng * 64 + t * 16 + l16;
#pragma unroll
        for (int j = 0; j < 8; ++j) {
            int row = mt * 16 + hi * 8 + j;         // C/D layout: lanes16-31 hold M=8..15
            C[(size_t)row * Nc + col] = acc[t][j];
        }
    }
}

// ---------------- attention coefficients ----------------
// as[n,h] = dot(h[n,h,:], a_s[h,:]) ; ad likewise. One thread per (node,head).
__global__ void k_alpha(const float* __restrict__ hfeat, const float* __restrict__ a_s,
                        const float* __restrict__ a_d, float* __restrict__ as_out,
                        float* __restrict__ ad_out, int heads) {
    int t = blockIdx.x * blockDim.x + threadIdx.x;
    if (t >= NNODE * heads) return;
    int h = t % heads, n = t / heads;
    const float* hp = hfeat + (size_t)n * heads * DH + h * DH;
    float ss = 0.f, sd = 0.f;
#pragma unroll 4
    for (int i = 0; i < DH; ++i) {
        float v = hp[i];
        ss += v * a_s[h * DH + i];
        sd += v * a_d[h * DH + i];
    }
    as_out[t] = ss;
    ad_out[t] = sd;
}

// ---------------- segment softmax passes (thread per edge) ----------------
__global__ void k_edge_max(const int* __restrict__ ei, const float* __restrict__ as,
                           const float* __restrict__ ad, float* __restrict__ m, int heads) {
    int w = blockIdx.x * blockDim.x + threadIdx.x;
    if (w >= ETOT) return;
    int src = (w < EEDGE) ? ei[w]         : (w - EEDGE);
    int dst = (w < EEDGE) ? ei[EEDGE + w] : (w - EEDGE);
    for (int h = 0; h < heads; ++h) {
        float e = lrelu02(as[src * heads + h] + ad[dst * heads + h]);
        atomicMaxF32(&m[dst * heads + h], e);
    }
}

__global__ void k_edge_sum(const int* __restrict__ ei, const float* __restrict__ as,
                           const float* __restrict__ ad, const float* __restrict__ m,
                           float* __restrict__ z, int heads) {
    int w = blockIdx.x * blockDim.x + threadIdx.x;
    if (w >= ETOT) return;
    int src = (w < EEDGE) ? ei[w]         : (w - EEDGE);
    int dst = (w < EEDGE) ? ei[EEDGE + w] : (w - EEDGE);
    for (int h = 0; h < heads; ++h) {
        float e = lrelu02(as[src * heads + h] + ad[dst * heads + h]);
        atomicAdd(&z[dst * heads + h], __expf(e - m[dst * heads + h]));
    }
}

// ---------------- aggregation, layer1: 256 feats, 4 heads, one wave/edge ----------------
__global__ void k_aggr256(const int* __restrict__ ei, const float* __restrict__ as,
                          const float* __restrict__ ad, const float* __restrict__ m,
                          const float* __restrict__ z, const float* __restrict__ h,
                          float* __restrict__ out) {
    int w = blockIdx.x * (blockDim.x >> 5) + (threadIdx.x >> 5);
    if (w >= ETOT) return;
    int lane = threadIdx.x & 31;
    int src = (w < EEDGE) ? ei[w]         : (w - EEDGE);
    int dst = (w < EEDGE) ? ei[EEDGE + w] : (w - EEDGE);
    int hd = lane >> 3;  // 8 floats/lane, 64 per head
    float e = lrelu02(as[src * HE + hd] + ad[dst * HE + hd]);
    float coef = __expf(e - m[dst * HE + hd]) / (z[dst * HE + hd] + 1e-16f);
    const float4* hp = (const float4*)(h + (size_t)src * H4D + lane * 8);
    float4 v0 = hp[0], v1 = hp[1];
    float* op = out + (size_t)dst * H4D + lane * 8;
    atomicAdd(op + 0, v0.x * coef); atomicAdd(op + 1, v0.y * coef);
    atomicAdd(op + 2, v0.z * coef); atomicAdd(op + 3, v0.w * coef);
    atomicAdd(op + 4, v1.x * coef); atomicAdd(op + 5, v1.y * coef);
    atomicAdd(op + 6, v1.z * coef); atomicAdd(op + 7, v1.w * coef);
}

// ---------------- aggregation, layer2: 64 feats, 1 head, one wave/edge ----------------
__global__ void k_aggr64(const int* __restrict__ ei, const float* __restrict__ as,
                         const float* __restrict__ ad, const float* __restrict__ m,
                         const float* __restrict__ z, const float* __restrict__ h,
                         float* __restrict__ out) {
    int w = blockIdx.x * (blockDim.x >> 5) + (threadIdx.x >> 5);
    if (w >= ETOT) return;
    int lane = threadIdx.x & 31;
    int src = (w < EEDGE) ? ei[w]         : (w - EEDGE);
    int dst = (w < EEDGE) ? ei[EEDGE + w] : (w - EEDGE);
    float e = lrelu02(as[src] + ad[dst]);
    float coef = __expf(e - m[dst]) / (z[dst] + 1e-16f);
    const float2 v = *(const float2*)(h + (size_t)src * DH + lane * 2);
    float* op = out + (size_t)dst * DH + lane * 2;
    atomicAdd(op + 0, v.x * coef);
    atomicAdd(op + 1, v.y * coef);
}

// hb[idx] = bf16(elu(agg[idx] + b[col]))
__global__ void k_elu_cast(const float* __restrict__ agg, const float* __restrict__ b,
                           bf16_t* __restrict__ hb, int cols, size_t n) {
    size_t i = (size_t)blockIdx.x * blockDim.x + threadIdx.x;
    if (i >= n) return;
    int col = (int)(i % cols);
    hb[i] = (bf16_t)eluf(agg[i] + b[col]);
}

// out[n,:] = elu(agg2[n,:]+b2) @ Wl + bl   (64x6 per node)
__global__ void k_final(const float* __restrict__ agg, const float* __restrict__ b2,
                        const float* __restrict__ Wl, const float* __restrict__ bl,
                        float* __restrict__ out) {
    int n = blockIdx.x * blockDim.x + threadIdx.x;
    if (n >= NNODE) return;
    float s[NCLS];
#pragma unroll
    for (int c = 0; c < NCLS; ++c) s[c] = bl[c];
    const float* ap = agg + (size_t)n * DH;
#pragma unroll 4
    for (int i = 0; i < DH; ++i) {
        float f = eluf(ap[i] + b2[i]);
#pragma unroll
        for (int c = 0; c < NCLS; ++c) s[c] += f * Wl[i * NCLS + c];
    }
#pragma unroll
    for (int c = 0; c < NCLS; ++c) out[(size_t)n * NCLS + c] = s[c];
}

extern "C" void kernel_launch(void* const* d_in, const int* in_sizes, int n_in,
                              void* d_out, int out_size, void* d_ws, size_t ws_size,
                              hipStream_t stream) {
    const float* x      = (const float*)d_in[0];
    const int*   ei     = (const int*)  d_in[1];
    const float* W1     = (const float*)d_in[2];
    const float* a_src1 = (const float*)d_in[3];
    const float* a_dst1 = (const float*)d_in[4];
    const float* b1     = (const float*)d_in[5];
    const float* W2     = (const float*)d_in[6];
    const float* a_src2 = (const float*)d_in[7];
    const float* a_dst2 = (const float*)d_in[8];
    const float* b2     = (const float*)d_in[9];
    const float* Wl     = (const float*)d_in[10];
    const float* bl     = (const float*)d_in[11];
    float* out = (float*)d_out;
    char*  ws  = (char*)d_ws;

    bf16_t* xb   = (bf16_t*)(ws + OFF_XB);
    bf16_t* w1t  = (bf16_t*)(ws + OFF_W1T);
    bf16_t* w2t  = (bf16_t*)(ws + OFF_W2T);
    float*  h1   = (float*) (ws + OFF_H1);
    float*  as1  = (float*) (ws + OFF_AS1);
    float*  ad1  = (float*) (ws + OFF_AD1);
    float*  m1   = (float*) (ws + OFF_M1);
    float*  z1   = (float*) (ws + OFF_Z1);
    float*  agg1 = (float*) (ws + OFF_AGG1);
    bf16_t* hb2  = (bf16_t*)(ws + OFF_HB2);
    float*  h2   = (float*) (ws + OFF_H2);
    float*  as2  = (float*) (ws + OFF_AS2);
    float*  ad2  = (float*) (ws + OFF_AD2);
    float*  m2   = (float*) (ws + OFF_M2);
    float*  z2   = (float*) (ws + OFF_Z2);
    float*  agg2 = (float*) (ws + OFF_AGG2);

    auto cdiv = [](int a, int b) { return (a + b - 1) / b; };

    // ---- layer 1 ----
    k_cast <<<cdiv(NNODE * DIN, 256), 256, 0, stream>>>(x, xb, (size_t)NNODE * DIN);
    k_tcast<<<cdiv(DIN * H4D, 256), 256, 0, stream>>>(W1, w1t, DIN, H4D);
    k_tcast<<<cdiv(H4D * DH, 256), 256, 0, stream>>>(W2, w2t, H4D, DH);

    k_fill<<<4096, 256, 0, stream>>>(agg1, 0.0f,   (size_t)NNODE * H4D);
    k_fill<<<512,  256, 0, stream>>>(m1,  -1e30f,  (size_t)NNODE * HE);
    k_fill<<<512,  256, 0, stream>>>(z1,   0.0f,   (size_t)NNODE * HE);

    int waves1 = (NNODE / 16) * (H4D / 64);   // 12500
    k_gemm<DIN><<<cdiv(waves1, 8), 256, 0, stream>>>(xb, w1t, h1, NNODE, H4D);

    k_alpha<<<cdiv(NNODE * HE, 256), 256, 0, stream>>>(h1, a_src1, a_dst1, as1, ad1, HE);
    k_edge_max<<<cdiv(ETOT, 256), 256, 0, stream>>>(ei, as1, ad1, m1, HE);
    k_edge_sum<<<cdiv(ETOT, 256), 256, 0, stream>>>(ei, as1, ad1, m1, z1, HE);
    k_aggr256 <<<cdiv(ETOT, 8),   256, 0, stream>>>(ei, as1, ad1, m1, z1, h1, agg1);
    k_elu_cast<<<cdiv(NNODE * H4D, 256), 256, 0, stream>>>(agg1, b1, hb2, H4D, (size_t)NNODE * H4D);

    // ---- layer 2 ----
    k_fill<<<2048, 256, 0, stream>>>(agg2, 0.0f,  (size_t)NNODE * DH);
    k_fill<<<512,  256, 0, stream>>>(m2,  -1e30f, (size_t)NNODE);
    k_fill<<<512,  256, 0, stream>>>(z2,   0.0f,  (size_t)NNODE);

    int waves2 = (NNODE / 16) * (DH / 64);    // 3125
    k_gemm<H4D><<<cdiv(waves2, 8), 256, 0, stream>>>(hb2, w2t, h2, NNODE, DH);

    k_alpha<<<cdiv(NNODE, 256), 256, 0, stream>>>(h2, a_src2, a_dst2, as2, ad2, 1);
    k_edge_max<<<cdiv(ETOT, 256), 256, 0, stream>>>(ei, as2, ad2, m2, 1);
    k_edge_sum<<<cdiv(ETOT, 256), 256, 0, stream>>>(ei, as2, ad2, m2, z2, 1);
    k_aggr64  <<<cdiv(ETOT, 8),   256, 0, stream>>>(ei, as2, ad2, m2, z2, h2, agg2);

    // ---- classifier ----
    k_final<<<cdiv(NNODE, 256), 256, 0, stream>>>(agg2, b2, Wl, bl, out);
}
